// WeightOnlyInt8Linear_90580860272696
// MI455X (gfx1250) — matrix operand order repaired and checked by reading, not compiled, not run on hardware
//
#include <hip/hip_runtime.h>

#define TOKENS 8192
#define D_IN   4096
#define D_OUT  4096

#define BM 128
#define BN 128
#define BK 64

typedef __attribute__((ext_vector_type(16))) _Float16 v16h;
typedef __attribute__((ext_vector_type(8)))  _Float16 v8h;
typedef __attribute__((ext_vector_type(4)))  _Float16 v4h;
typedef __attribute__((ext_vector_type(8)))  float    v8f;
typedef __attribute__((ext_vector_type(4)))  float    v4f;
typedef __attribute__((ext_vector_type(4)))  int      v4i;

__global__ __launch_bounds__(256)
void WeightOnlyInt8Linear_wmma_kernel(const float* __restrict__ A,
                                      const int*   __restrict__ W,
                                      const float* __restrict__ scales,
                                      float*       __restrict__ out)
{
    // Double-buffered f16 tiles: 2 * (128*64 + 128*64) * 2B = 64 KB of 320 KB LDS.
    __shared__ alignas(64) _Float16 lA[2][BM * BK];
    __shared__ alignas(64) _Float16 lB[2][BN * BK];

    const int tid   = threadIdx.x;
    const int lane  = tid & 31;
    const int wave  = tid >> 5;        // 0..7
    const int waveM = wave >> 2;       // 0..1  -> 64-row strip
    const int waveN = wave & 3;        // 0..3  -> 32-col strip
    const int g     = lane >> 4;       // lane half (WMMA K-group)
    const int ln15  = lane & 15;

    const int tileM = blockIdx.y * BM;
    const int tileN = blockIdx.x * BN;

    v8f acc[4][2] = {};                // eight 16x16 f32 accumulator tiles per wave

    // Register staging for the next K-tile (raw, converted during LDS store
    // so the loads stay outstanding across the WMMA block).
    v4f aReg[8];
    v4i bReg[8];

    auto loadGlobal = [&](int kt) {
        #pragma unroll
        for (int i = 0; i < 8; ++i) {
            int c  = tid + i * 256;          // 0..2047 chunks of 4 elems
            int r  = c >> 4;                 // 0..127 tile row
            int k4 = (c & 15) << 2;          // 0..60  k offset
            aReg[i] = *(const v4f*)(A + (size_t)(tileM + r) * D_IN + kt + k4);
            bReg[i] = *(const v4i*)(W + (size_t)(tileN + r) * D_IN + kt + k4);
        }
    };

    auto storeLDS = [&](int buf) {
        #pragma unroll
        for (int i = 0; i < 8; ++i) {
            int c  = tid + i * 256;
            int r  = c >> 4;
            int k4 = (c & 15) << 2;
            v4h ha, hb;
            #pragma unroll
            for (int e = 0; e < 4; ++e) {
                ha[e] = (_Float16)aReg[i][e];
                hb[e] = (_Float16)(float)bReg[i][e];   // int8 values exact in f16
            }
            *(v4h*)&lA[buf][r * BK + k4] = ha;
            *(v4h*)&lB[buf][r * BK + k4] = hb;
        }
    };

    loadGlobal(0);
    storeLDS(0);
    __syncthreads();

    for (int kt = 0; kt < D_IN; kt += BK) {
        const int  buf     = (kt / BK) & 1;
        const bool hasNext = (kt + BK) < D_IN;

        if (hasNext) {
            if (kt + 2 * BK < D_IN) {
                // L2 warm-up two K-tiles ahead (global_prefetch_b8)
                int r = tid >> 4, k4 = (tid & 15) << 2;
                __builtin_prefetch(A + (size_t)(tileM + r) * D_IN + kt + 2 * BK + k4, 0, 1);
                __builtin_prefetch(W + (size_t)(tileN + r) * D_IN + kt + 2 * BK + k4, 0, 1);
            }
            loadGlobal(kt + BK);           // overlaps with WMMA block below
        }

        #pragma unroll
        for (int kk = 0; kk < BK; kk += 32) {
            // B fragments: lane = column, lane-group g holds K = g*16..g*16+15 (contiguous)
            v16h bf[2];
            #pragma unroll
            for (int tn = 0; tn < 2; ++tn) {
                int col = waveN * 32 + tn * 16 + ln15;
                bf[tn] = *(const v16h*)&lB[buf][col * BK + kk + g * 16];
            }
            // A fragments: lane = row, group g holds K = g*16..g*16+15 (contiguous)
            #pragma unroll
            for (int tm = 0; tm < 4; ++tm) {
                int row = waveM * 64 + tm * 16 + ln15;
                const _Float16* base = &lA[buf][row * BK + kk];
                v8h lo = *(const v8h*)(base + g * 16);
                v8h hi = *(const v8h*)(base + g * 16 + 8);
                v16h af = __builtin_shufflevector(lo, hi,
                            0,1,2,3,4,5,6,7,8,9,10,11,12,13,14,15);
                #pragma unroll
                for (int tn = 0; tn < 2; ++tn) {
                    acc[tm][tn] = __builtin_amdgcn_wmma_f32_16x16x32_f16(
                        false, af, false, bf[tn],
                        (short)0, acc[tm][tn], false, false);
                }
            }
        }

        if (hasNext) storeLDS(buf ^ 1);    // writes the buffer not being read
        __syncthreads();
    }

    // Epilogue: C layout is VGPR v, lane-group g -> M = v + 8*g, N = lane&15.
    #pragma unroll
    for (int tn = 0; tn < 2; ++tn) {
        int col = tileN + waveN * 32 + tn * 16 + ln15;
        float s = scales[col];
        #pragma unroll
        for (int tm = 0; tm < 4; ++tm) {
            int row0 = tileM + waveM * 64 + tm * 16 + g * 8;
            #pragma unroll
            for (int v = 0; v < 8; ++v) {
                out[(size_t)(row0 + v) * D_OUT + col] = acc[tm][tn][v] * s;
            }
        }
    }
}

extern "C" void kernel_launch(void* const* d_in, const int* in_sizes, int n_in,
                              void* d_out, int out_size, void* d_ws, size_t ws_size,
                              hipStream_t stream) {
    const float* A = (const float*)d_in[0];   // input  [8192, 4096] fp32
    const int*   W = (const int*)d_in[1];     // weight [4096, 4096] int8-in-int32
    const float* S = (const float*)d_in[2];   // scales [4096] fp32
    float*       O = (float*)d_out;           // out    [8192, 4096] fp32

    dim3 grid(D_OUT / BN, TOKENS / BM);       // 32 x 64 workgroups
    WeightOnlyInt8Linear_wmma_kernel<<<grid, 256, 0, stream>>>(A, W, S, O);
}
